// NodeWiseMamba_77352361001429
// MI455X (gfx1250) — compile-verified, matrix operand order checked by reading, not run on hardware
//
#include <hip/hip_runtime.h>

#define NB 8
#define TT 10
#define HH 256
#define SS 16
#define RR 16
#define DCC 4
#define ROWS (NB*TT)   /* 80 = 5 m-tiles exactly */
#define NWAVES 8

typedef __attribute__((ext_vector_type(16))) __bf16 v16bf;
typedef __attribute__((ext_vector_type(8)))  float  v8f;
typedef __attribute__((ext_vector_type(4)))  unsigned int v4u;

union FragU { unsigned int u[8]; v4u q[2]; v16bf v; };

__device__ __forceinline__ unsigned short f2bf(float f){
  unsigned int u = __float_as_uint(f);
  unsigned int r = u + 0x7FFFu + ((u >> 16) & 1u);
  return (unsigned short)(r >> 16);
}
__device__ __forceinline__ float bf2f(unsigned short h){
  return __uint_as_float(((unsigned int)h) << 16);
}
__device__ __forceinline__ unsigned int pack2bf(float a, float b){
  return (unsigned int)f2bf(a) | ((unsigned int)f2bf(b) << 16);
}
// (lane,dword) -> local K index for 16-bit WMMA A/B fragments (ISA 7.12.2)
__device__ __forceinline__ int kloc(int lane, int d){
  return ((d & 3) << 1) + ((d & 4) << 2) + ((lane >> 4) << 3);
}
__device__ __forceinline__ float wred(float s){
  s += __shfl_xor(s, 16, 32); s += __shfl_xor(s, 8, 32);
  s += __shfl_xor(s, 4, 32);  s += __shfl_xor(s, 2, 32);
  s += __shfl_xor(s, 1, 32);  return s;
}
__device__ __forceinline__ float gelu_f(float x){ return 0.5f*x*(1.0f + erff(x*0.7071067811865476f)); }
__device__ __forceinline__ float silu_f(float x){ return x/(1.0f + __expf(-x)); }

// ---------------- weight pre-pack: f32 (K,N) row-major -> bf16 WMMA B-fragment order
// dst layout: [kt][nt][lane(32)][dword(8)], each dword = bf16 pair (k, k+1)
__global__ void pack_w(const float* __restrict__ src, unsigned int* __restrict__ dst,
                       int K, int N, int ld){
  int idx = blockIdx.x*blockDim.x + threadIdx.x;
  int NT = N >> 4;
  int KT = (K + 31) >> 5;
  int total = KT*NT*32;
  if (idx >= total) return;
  int lane = idx & 31;
  int tile = idx >> 5;
  int nt = tile % NT;
  int kt = tile / NT;
  int n  = nt*16 + (lane & 15);
  unsigned int* outp = dst + (((size_t)tile)*32 + lane)*8;
  #pragma unroll
  for (int d = 0; d < 8; ++d){
    int k = kt*32 + kloc(lane, d);
    float a = (k     < K) ? src[(size_t)k    *ld + n] : 0.0f;
    float b = (k + 1 < K) ? src[(size_t)(k+1)*ld + n] : 0.0f;
    outp[d] = pack2bf(a, b);
  }
}

// ---------------- generic fused-block GEMM: (MT*16 x K) @ (K x NT*16), bf16 WMMA, f32 acc
template<int MT, bool ABF16>
__device__ void gemm_wmma(const unsigned short* Ab, const float* Af, int As, int Kdim,
                          const unsigned int* __restrict__ Bpk, int NT, int KT,
                          float* Out, int Os,
                          const float* Cacc, int Cs,
                          const float* __restrict__ bias){
  const int wave = threadIdx.x >> 5;
  const int lane = threadIdx.x & 31;
  const int lr   = lane & 15;
  const int hi8  = (lane >> 4) << 3;     // 0 or 8
  for (int nt = wave; nt < NT; nt += NWAVES){
    v8f acc[MT];
    #pragma unroll
    for (int mt = 0; mt < MT; ++mt){
      if (Cacc){
        #pragma unroll
        for (int g = 0; g < 8; ++g){
          int row = mt*16 + g + hi8;
          acc[mt][g] = Cacc[row*Cs + nt*16 + lr];
        }
      } else {
        #pragma unroll
        for (int g = 0; g < 8; ++g) acc[mt][g] = 0.0f;
      }
    }
    for (int kt = 0; kt < KT; ++kt){
      FragU bu;
      const v4u* bp = (const v4u*)(Bpk + ((((size_t)kt*NT + nt)*32 + lane) << 3));
      bu.q[0] = bp[0];
      bu.q[1] = bp[1];
      #pragma unroll
      for (int mt = 0; mt < MT; ++mt){
        FragU au;
        int m = mt*16 + lr;
        if constexpr (ABF16){
          // A fragment = two contiguous 16B runs per lane (k_local {0..7|8..15} and {16..23|24..31})
          const unsigned short* ap = Ab + m*As + kt*32 + hi8;
          au.q[0] = *(const v4u*)(ap);
          au.q[1] = *(const v4u*)(ap + 16);
        } else {
          #pragma unroll
          for (int d = 0; d < 8; ++d){
            int k = kt*32 + kloc(lane, d);
            float a = (k     < Kdim) ? Af[m*As + k    ] : 0.0f;
            float b = (k + 1 < Kdim) ? Af[m*As + k + 1] : 0.0f;
            au.u[d] = pack2bf(a, b);
          }
        }
        acc[mt] = __builtin_amdgcn_wmma_f32_16x16x32_bf16(
            false, au.v, false, bu.v, (short)0, acc[mt], false, false);
      }
    }
    float bv = bias ? bias[nt*16 + lr] : 0.0f;
    #pragma unroll
    for (int mt = 0; mt < MT; ++mt){
      #pragma unroll
      for (int g = 0; g < 8; ++g){
        int row = mt*16 + g + hi8;
        Out[row*Os + nt*16 + lr] = acc[mt][g] + bv;
      }
    }
  }
}

// ---------------- fully fused network: 8 nodes per block, everything in LDS
__global__ __launch_bounds__(256) void fused_mamba(
    const float* __restrict__ x_hist,
    const float* __restrict__ in_proj_b,
    const float* __restrict__ rms_in_w,
    const float* __restrict__ pre_w,
    const float* __restrict__ conv_w,
    const float* __restrict__ conv_b,
    const float* __restrict__ dt_bias,
    const float* __restrict__ A_log,
    const float* __restrict__ Dp,
    const float* __restrict__ final_w,
    const float* __restrict__ out_proj_b,
    const unsigned int* __restrict__ pk_in,
    const unsigned int* __restrict__ pk_min,
    const unsigned int* __restrict__ pk_xp,
    const unsigned int* __restrict__ pk_dtp,
    const unsigned int* __restrict__ pk_outw,
    const unsigned int* __restrict__ pk_oproj,
    float* __restrict__ out){
  extern __shared__ char smem[];
  float* ubuf  = (float*)smem;                         // [80][512] GEMM outs: u||z, later dt||z
  float* xbuf  = ubuf + ROWS*512;                      // [80][256] residual (f32)
  float* xdbuf = xbuf + ROWS*HH;                       // [80][48]  x_dbl
  int*   rflag = (int*)(xdbuf + ROWS*48);              // [80] per-row mask
  int*   lastx = rflag + ROWS;                         // [8]  last valid t per node
  unsigned short* tbuf = (unsigned short*)(lastx + NB);// [80][256] bf16 GEMM-A staging (16B aligned)
  unsigned short* fbuf = tbuf + ROWS*HH;               // [16][256] final gathered rows

  const int tid  = threadIdx.x;
  const int wave = tid >> 5, lane = tid & 31;
  const int node0 = blockIdx.x * NB;

  // ---- Stage 0: load x_hist -> bf16 tbuf; per-row |x| flags for mask
  for (int r = wave; r < ROWS; r += NWAVES){
    const float* src = x_hist + ((size_t)node0*TT + r)*HH;
    float s = 0.0f;
    #pragma unroll
    for (int j = 0; j < 8; ++j){
      float v = src[lane + j*32];
      tbuf[r*HH + lane + j*32] = f2bf(v);
      s += fabsf(v);
    }
    s = wred(s);
    if (lane == 0) rflag[r] = (s > 0.0f) ? 1 : 0;
  }
  __syncthreads();
  if (tid < NB){
    int c = 0;
    for (int t = 0; t < TT; ++t) c += rflag[tid*TT + t];
    if (c < 1) c = 1;
    lastx[tid] = c - 1;
  }
  __syncthreads();

  // ---- in_proj GEMM (+bias) -> ubuf [80][256]
  gemm_wmma<5, true>(tbuf, nullptr, HH, HH, pk_in, 16, 8, ubuf, HH, nullptr, 0, in_proj_b);
  __syncthreads();

  // ---- x = gelu(rmsnorm(., rms_in_w)) -> xbuf (residual)
  for (int r = wave; r < ROWS; r += NWAVES){
    float vals[8]; float s = 0.0f;
    #pragma unroll
    for (int j = 0; j < 8; ++j){ float v = ubuf[r*HH + lane + j*32]; vals[j] = v; s += v*v; }
    s = wred(s);
    float inv = rsqrtf(s*(1.0f/HH) + 1e-5f);
    #pragma unroll
    for (int j = 0; j < 8; ++j){
      int h = lane + j*32;
      xbuf[r*HH + h] = gelu_f(vals[j]*inv*rms_in_w[h]);
    }
  }
  __syncthreads();

  // ---- 2 mamba layers
  for (int l = 0; l < 2; ++l){
    // rmsnorm(resid, pre_w) -> tbuf bf16
    for (int r = wave; r < ROWS; r += NWAVES){
      float vals[8]; float s = 0.0f;
      #pragma unroll
      for (int j = 0; j < 8; ++j){ float v = xbuf[r*HH + lane + j*32]; vals[j] = v; s += v*v; }
      s = wred(s);
      float inv = rsqrtf(s*(1.0f/HH) + 1e-5f);
      #pragma unroll
      for (int j = 0; j < 8; ++j){
        int h = lane + j*32;
        tbuf[r*HH + h] = f2bf(vals[j]*inv*pre_w[l*HH + h]);
      }
    }
    __syncthreads();
    // u||z = t @ m_in_w  -> ubuf [80][512]
    gemm_wmma<5, true>(tbuf, nullptr, HH, HH, pk_min + l*65536, 32, 8, ubuf, 512, nullptr, 0, nullptr);
    __syncthreads();
    // causal depthwise conv (DC=4) + silu -> tbuf bf16
    for (int p = tid; p < NB*HH; p += 256){
      int nl = p >> 8, h = p & 255;
      float cw0 = conv_w[(l*HH + h)*DCC + 0];
      float cw1 = conv_w[(l*HH + h)*DCC + 1];
      float cw2 = conv_w[(l*HH + h)*DCC + 2];
      float cw3 = conv_w[(l*HH + h)*DCC + 3];
      float cb  = conv_b[l*HH + h];
      float w0 = 0.0f, w1 = 0.0f, w2 = 0.0f;
      for (int t = 0; t < TT; ++t){
        int row = nl*TT + t;
        float ut = ubuf[row*512 + h];
        float uc = cb + cw0*w0 + cw1*w1 + cw2*w2 + cw3*ut;
        w0 = w1; w1 = w2; w2 = ut;
        tbuf[row*HH + h] = f2bf(silu_f(uc));
      }
    }
    __syncthreads();
    // x_dbl = u @ xproj (48 cols) -> xdbuf
    gemm_wmma<5, true>(tbuf, nullptr, HH, HH, pk_xp + l*6144, 3, 8, xdbuf, 48, nullptr, 0, nullptr);
    __syncthreads();
    // dt_lin = x_dbl[:, :16] @ dtproj + dt_bias -> ubuf cols [0,256)  (K padded 16->32)
    gemm_wmma<5, false>(nullptr, xdbuf, 48, RR, pk_dtp + l*4096, 16, 1, ubuf, 512, nullptr, 0, dt_bias + l*HH);
    __syncthreads();
    // selective scan over T, state (S=16) in registers; writes g -> tbuf bf16
    for (int p = tid; p < NB*HH; p += 256){
      int nl = p >> 8, h = p & 255;
      float Av[SS];
      #pragma unroll
      for (int s = 0; s < SS; ++s) Av[s] = -__expf(A_log[(l*HH + h)*SS + s]);
      float Dv = Dp[l*HH + h];
      float st[SS];
      #pragma unroll
      for (int s = 0; s < SS; ++s) st[s] = 0.0f;
      for (int t = 0; t < TT; ++t){
        int row = nl*TT + t;
        float dtr = ubuf[row*512 + h];
        float dtv = (dtr > 20.0f) ? dtr : log1pf(__expf(dtr));   // softplus
        float ut  = bf2f(tbuf[row*HH + h]);
        float du  = dtv * ut;
        float y = 0.0f;
        #pragma unroll
        for (int s = 0; s < SS; ++s){
          float Bs = xdbuf[row*48 + RR + s];
          float Cs = xdbuf[row*48 + RR + SS + s];
          st[s] = __expf(dtv*Av[s])*st[s] + du*Bs;
          y += st[s]*Cs;
        }
        float zv = ubuf[row*512 + 256 + h];
        float g  = (y + ut*Dv) * silu_f(zv);
        tbuf[row*HH + h] = f2bf(g);
      }
    }
    __syncthreads();
    // resid += g @ out_w   (C operand = current residual)
    gemm_wmma<5, true>(tbuf, nullptr, HH, HH, pk_outw + l*32768, 16, 8, xbuf, HH, xbuf, HH, nullptr);
    __syncthreads();
  }

  // ---- final rmsnorm -> tbuf bf16
  for (int r = wave; r < ROWS; r += NWAVES){
    float vals[8]; float s = 0.0f;
    #pragma unroll
    for (int j = 0; j < 8; ++j){ float v = xbuf[r*HH + lane + j*32]; vals[j] = v; s += v*v; }
    s = wred(s);
    float inv = rsqrtf(s*(1.0f/HH) + 1e-5f);
    #pragma unroll
    for (int j = 0; j < 8; ++j){
      int h = lane + j*32;
      tbuf[r*HH + h] = f2bf(vals[j]*inv*final_w[h]);
    }
  }
  __syncthreads();
  // ---- gather last valid timestep per node into one 16-row tile
  for (int p = tid; p < 16*HH; p += 256){
    int rr = p >> 8, c = p & 255;
    fbuf[p] = (rr < NB) ? tbuf[(rr*TT + lastx[rr])*HH + c] : (unsigned short)0;
  }
  __syncthreads();
  // ---- out_proj GEMM (+bias) -> ubuf rows [0,16)
  gemm_wmma<1, true>(fbuf, nullptr, HH, HH, pk_oproj, 16, 8, ubuf, HH, nullptr, 0, out_proj_b);
  __syncthreads();
  // ---- gelu + store
  for (int p = tid; p < NB*HH; p += 256){
    out[(size_t)(node0 + (p >> 8))*HH + (p & 255)] = gelu_f(ubuf[p]);
  }
}

extern "C" void kernel_launch(void* const* d_in, const int* in_sizes, int n_in,
                              void* d_out, int out_size, void* d_ws, size_t ws_size,
                              hipStream_t stream){
  (void)in_sizes; (void)n_in; (void)out_size; (void)ws_size;
  const float* x_hist   = (const float*)d_in[0];
  const float* in_w     = (const float*)d_in[1];
  const float* in_b     = (const float*)d_in[2];
  const float* rms_in_w = (const float*)d_in[3];
  const float* pre_wp   = (const float*)d_in[4];
  const float* m_in_w   = (const float*)d_in[5];
  const float* m_conv_w = (const float*)d_in[6];
  const float* m_conv_b = (const float*)d_in[7];
  const float* m_xp_w   = (const float*)d_in[8];
  const float* m_dtp_w  = (const float*)d_in[9];
  const float* m_dt_b   = (const float*)d_in[10];
  const float* m_A_log  = (const float*)d_in[11];
  const float* m_D      = (const float*)d_in[12];
  const float* m_out_w  = (const float*)d_in[13];
  const float* final_w  = (const float*)d_in[14];
  const float* op_w     = (const float*)d_in[15];
  const float* op_b     = (const float*)d_in[16];

  unsigned int* ws      = (unsigned int*)d_ws;
  unsigned int* pk_in   = ws;                   // 8*16*256  = 32768 dwords
  unsigned int* pk_min  = pk_in  + 32768;       // 2 * 65536
  unsigned int* pk_xp   = pk_min + 2*65536;     // 2 * 6144
  unsigned int* pk_dtp  = pk_xp  + 2*6144;      // 2 * 4096
  unsigned int* pk_outw = pk_dtp + 2*4096;      // 2 * 32768
  unsigned int* pk_op   = pk_outw + 2*32768;    // 32768

  auto packs = [&](const float* src, unsigned int* dst, int K, int N){
    int KT = (K + 31)/32, NT = N/16;
    int total = KT*NT*32;
    pack_w<<<(total + 255)/256, 256, 0, stream>>>(src, dst, K, N, N);
  };
  packs(in_w,               pk_in,            256, 256);
  packs(m_in_w,             pk_min,           256, 512);
  packs(m_in_w + 256*512,   pk_min + 65536,   256, 512);
  packs(m_xp_w,             pk_xp,            256, 48);
  packs(m_xp_w + 256*48,    pk_xp + 6144,     256, 48);
  packs(m_dtp_w,            pk_dtp,           16,  256);
  packs(m_dtp_w + 16*256,   pk_dtp + 4096,    16,  256);
  packs(m_out_w,            pk_outw,          256, 256);
  packs(m_out_w + 256*256,  pk_outw + 32768,  256, 256);
  packs(op_w,               pk_op,            256, 256);

  size_t shmem = (size_t)(ROWS*512 + ROWS*HH + ROWS*48 + ROWS + NB)*4
               + (size_t)(ROWS*HH + 16*HH)*2;   // ~303 KB, under 320 KB/WGP
  fused_mamba<<<20000/NB, 256, shmem, stream>>>(
      x_hist, in_b, rms_in_w, pre_wp, m_conv_w, m_conv_b, m_dt_b, m_A_log,
      m_D, final_w, op_b,
      pk_in, pk_min, pk_xp, pk_dtp, pk_outw, pk_op,
      (float*)d_out);
}
